// SinkhornRouter_50818053046522
// MI455X (gfx1250) — compile-verified, matrix operand order checked by reading, not run on hardware
//
#include <hip/hip_runtime.h>
#include <math.h>

// ---------------- problem constants (from reference) ----------------
constexpr int Bc   = 4;
constexpr int Nc   = 8192;
constexpr int Dc   = 512;
constexpr int Ec   = 8;
constexpr int OUTc = 512;
constexpr int Mc   = Nc / Ec;   // 1024 tokens per expert
constexpr float EPSc = 1e-6f;

// GEMM tiling
constexpr int TM = 128, TN = 128, TK = 32;
constexpr int LDST = 40;        // padded LDS row stride (elems) to spread banks

// ---------------- WMMA types (gfx1250, wave32) ----------------
typedef __attribute__((ext_vector_type(16))) __bf16 v16bf;
typedef __attribute__((ext_vector_type(8)))  float  v8f;

union FragU { unsigned int u[8]; v16bf v; };

// CDNA5 async global->LDS path availability (probe via __has_builtin; fallback = reg staging)
#if defined(__AMDGCN__) && __has_builtin(__builtin_amdgcn_global_load_async_to_lds_b128)
#define USE_ASYNC_LDS 1
#else
#define USE_ASYNC_LDS 0
#endif

__device__ __forceinline__ void wait_asynccnt0() {
#if __has_builtin(__builtin_amdgcn_s_wait_asynccnt)
    __builtin_amdgcn_s_wait_asynccnt(0);
#else
    asm volatile("s_wait_asynccnt 0" ::: "memory");
#endif
}

#if USE_ASYNC_LDS
// builtin signature (from probe diagnostic): param0 = int4 in addrspace(1) (global src),
// param1 = int4 in addrspace(3) (LDS dst), then imm offset, imm cpol.
typedef int v4i_ __attribute__((vector_size(16)));
typedef v4i_ __attribute__((address_space(1))) gv4i;
typedef v4i_ __attribute__((address_space(3))) lv4i;

// one 16-byte async copy global -> LDS (tracked by ASYNCcnt, no VGPR round trip)
__device__ __forceinline__ void async_cp16(const unsigned short* src, unsigned short* dst) {
    __builtin_amdgcn_global_load_async_to_lds_b128((gv4i*)src, (lv4i*)dst, 0, 0);
}
#endif

// float -> bf16 bits, round-to-nearest-even (pure integer math)
__device__ __forceinline__ unsigned short f2bf(float f) {
    unsigned int u = __float_as_uint(f);
    unsigned int rnd = 0x7FFFu + ((u >> 16) & 1u);
    return (unsigned short)((u + rnd) >> 16);
}

// Load one 16x32 bf16 WMMA operand from an LDS tile laid out [rows][strideElems].
// ISA 7.12.2 (16-bit A/B): lane(0-15) row=lane, K={0..7,16..23}; lane(16-31) same row,
// K={8..15,24..31}; dword v holds the K pair at kstart(v).
__device__ __forceinline__ v16bf frag_ld(const unsigned short* rowbase, int strideElems,
                                         int r, int half) {
    FragU f;
    const unsigned short* p = rowbase + r * strideElems;
#pragma unroll
    for (int v = 0; v < 8; ++v) {
        int kstart = ((v < 4) ? (2 * v) : (16 + 2 * (v - 4))) + (half ? 8 : 0);
        f.u[v] = *reinterpret_cast<const unsigned int*>(p + kstart);
    }
    return f.v;
}

// ---------------- 0) zero the output ----------------
__global__ void moe_zero_f4(float4* p, int n4) {
    for (int i = blockIdx.x * blockDim.x + threadIdx.x; i < n4; i += gridDim.x * blockDim.x)
        p[i] = float4{0.f, 0.f, 0.f, 0.f};
}

// ---------------- 1) gate logits + x->bf16 conversion (fused) ----------------
__global__ void moe_gate_kernel(const float* __restrict__ x, const float* __restrict__ gw,
                                float* __restrict__ t, unsigned short* __restrict__ xb) {
    __shared__ float xs[Dc];
    __shared__ float part[128];
    const int tid = threadIdx.x;
    const size_t row = blockIdx.x;               // b*N + n
    float4 v = reinterpret_cast<const float4*>(x + row * Dc)[tid];
    xs[tid * 4 + 0] = v.x; xs[tid * 4 + 1] = v.y;
    xs[tid * 4 + 2] = v.z; xs[tid * 4 + 3] = v.w;
    unsigned short* xr = xb + row * Dc + tid * 4;
    xr[0] = f2bf(v.x); xr[1] = f2bf(v.y); xr[2] = f2bf(v.z); xr[3] = f2bf(v.w);
    __syncthreads();
    const int e = tid & 7, seg = tid >> 3;       // 16 segments x 32 elems
    float acc = 0.f;
    const int base = seg * 32;
#pragma unroll 8
    for (int k = 0; k < 32; ++k)
        acc += xs[base + k] * gw[(base + k) * Ec + e];
    part[tid] = acc;
    __syncthreads();
    for (int s = 64; s >= 8; s >>= 1) {
        if (tid < s) part[tid] += part[tid + s];
        __syncthreads();
    }
    if (tid < 8)
        t[row * Ec + tid] = logf(fmaxf(part[tid], EPSc));   // TEMPERATURE = 1
}

// ---------------- 2) experts -> bf16, transposed to [e][out][d] ----------------
__global__ void moe_wconv_kernel(const float* __restrict__ ex, unsigned short* __restrict__ wtb) {
    int gid = blockIdx.x * blockDim.x + threadIdx.x;
    if (gid >= Ec * Dc * OUTc) return;
    int o = gid % OUTc;
    int d = (gid / OUTc) % Dc;
    int e = gid / (OUTc * Dc);
    wtb[((size_t)e * OUTc + o) * Dc + d] = f2bf(ex[gid]);
}

// ---------------- 3a) sinkhorn column logsumexp over N per (b,e) ----------------
__global__ void moe_col_lse(const float* __restrict__ t, float* __restrict__ collse) {
    __shared__ float red[256];
    const int be = blockIdx.x;
    const int b = be / Ec, e = be % Ec;
    const float* base = t + (size_t)b * Nc * Ec + e;
    float mx = -INFINITY;
    for (int i = threadIdx.x; i < Nc; i += 256)
        mx = fmaxf(mx, base[(size_t)i * Ec]);
    red[threadIdx.x] = mx; __syncthreads();
    for (int s = 128; s >= 1; s >>= 1) {
        if (threadIdx.x < s) red[threadIdx.x] = fmaxf(red[threadIdx.x], red[threadIdx.x + s]);
        __syncthreads();
    }
    mx = red[0]; __syncthreads();
    float sm = 0.f;
    for (int i = threadIdx.x; i < Nc; i += 256)
        sm += expf(base[(size_t)i * Ec] - mx);
    red[threadIdx.x] = sm; __syncthreads();
    for (int s = 128; s >= 1; s >>= 1) {
        if (threadIdx.x < s) red[threadIdx.x] += red[threadIdx.x + s];
        __syncthreads();
    }
    if (threadIdx.x == 0) collse[be] = mx + logf(red[0]);
}

// ---------------- 3b) subtract col lse, row logsumexp over E; optional final exp --------
__global__ void moe_row_lse(float* __restrict__ t, const float* __restrict__ collse, int do_exp) {
    int gid = blockIdx.x * blockDim.x + threadIdx.x;
    if (gid >= Bc * Nc) return;
    const int b = gid / Nc;
    float4 lo = reinterpret_cast<float4*>(t)[gid * 2];
    float4 hi = reinterpret_cast<float4*>(t)[gid * 2 + 1];
    float v[8] = {lo.x, lo.y, lo.z, lo.w, hi.x, hi.y, hi.z, hi.w};
    float mx = -INFINITY;
#pragma unroll
    for (int e = 0; e < 8; ++e) { v[e] -= collse[b * Ec + e]; mx = fmaxf(mx, v[e]); }
    float sm = 0.f;
#pragma unroll
    for (int e = 0; e < 8; ++e) sm += expf(v[e] - mx);
    float lse = mx + logf(sm);
#pragma unroll
    for (int e = 0; e < 8; ++e) {
        float r = v[e] - lse;
        v[e] = do_exp ? expf(r) : r;
    }
    reinterpret_cast<float4*>(t)[gid * 2]     = float4{v[0], v[1], v[2], v[3]};
    reinterpret_cast<float4*>(t)[gid * 2 + 1] = float4{v[4], v[5], v[6], v[7]};
}

// ---------------- 4) top-k (k=1024 of 8192) per (b,e): in-LDS bitonic sort ----------------
__global__ __launch_bounds__(1024) void moe_topk(const float* __restrict__ gates,
                                                 int* __restrict__ idxb,
                                                 float* __restrict__ valb) {
    __shared__ float sk[Nc];
    __shared__ int   si[Nc];
    const int be = blockIdx.x;
    const int b = be / Ec, e = be % Ec;
    for (int i = threadIdx.x; i < Nc; i += 1024) {
        sk[i] = gates[((size_t)b * Nc + i) * Ec + e];
        si[i] = i;
    }
    __syncthreads();
    for (int ksz = 2; ksz <= Nc; ksz <<= 1) {
        for (int jsz = ksz >> 1; jsz > 0; jsz >>= 1) {
            for (int i = threadIdx.x; i < Nc; i += 1024) {
                int ixj = i ^ jsz;
                if (ixj > i) {
                    bool desc = ((i & ksz) == 0);
                    float k1 = sk[i], k2 = sk[ixj];
                    int   i1 = si[i], i2 = si[ixj];
                    bool firstGreater = (k1 > k2) || (k1 == k2 && i1 < i2);
                    bool doSwap = desc ? !firstGreater : firstGreater;
                    if (doSwap) { sk[i] = k2; sk[ixj] = k1; si[i] = i2; si[ixj] = i1; }
                }
            }
            __syncthreads();
        }
    }
    if (threadIdx.x < Mc) {
        idxb[(size_t)be * Mc + threadIdx.x] = si[threadIdx.x];
        valb[(size_t)be * Mc + threadIdx.x] = sk[threadIdx.x];
    }
}

// ---------------- 5) ownership (last-writer-wins over flat e*m order) ----------------
__global__ void moe_owner_init(int* owner, int n) {
    int i = blockIdx.x * blockDim.x + threadIdx.x;
    if (i < n) owner[i] = -1;
}
__global__ void moe_owner_fill(const int* __restrict__ idxb, int* __restrict__ owner) {
    int gid = blockIdx.x * blockDim.x + threadIdx.x;
    if (gid >= Bc * Ec * Mc) return;
    int b = gid / (Ec * Mc);
    int j = gid % (Ec * Mc);
    int token = idxb[gid];
    atomicMax(&owner[(size_t)b * Nc + token], j);
}

// ---------------- 6) expert GEMM (bf16 WMMA), double-buffered, async LDS fill -----------
// grid: (OUT/TN, Mc/TM, E*B); block: 256 (8 waves). Block tile 128x128.
// Waves: 4 (M) x 2 (N); each wave computes 32(M) x 64(N) = 2 A-frags x 4 B-frags = 8 WMMA/K-step.
__global__ __launch_bounds__(256) void moe_gemm_scatter(
        const unsigned short* __restrict__ xb,   // bf16 x [B][N][D]
        const unsigned short* __restrict__ wtb,  // bf16 W^T [E][OUT][D]
        const int*   __restrict__ idxb,          // [B][E][Mc]
        const float* __restrict__ valb,          // [B][E][Mc]
        const int*   __restrict__ owner,         // [B][N]
        float* __restrict__ out) {               // [B][N][OUT]
    __shared__ __align__(16) unsigned short Asm[2][TM][LDST];
    __shared__ __align__(16) unsigned short Bsm[2][TN][LDST];

    const int tid = threadIdx.x;
    const int e = blockIdx.z / Bc;
    const int b = blockIdx.z % Bc;
    const int nbase = blockIdx.x * TN;
    const int mbase = blockIdx.y * TM;
    const int beBase = (b * Ec + e) * Mc;

    // loaders: each matrix tile = TM x TK elems = 512 chunks of 16B; 2 chunks/thread.
    const int cA0 = tid,        cA1 = tid + 256;      // chunk ids
    const int rA0 = cA0 >> 2,   kA0 = (cA0 & 3) * 8;
    const int rA1 = cA1 >> 2,   kA1 = (cA1 & 3) * 8;
    const int tok0 = idxb[beBase + mbase + rA0];
    const int tok1 = idxb[beBase + mbase + rA1];
    const unsigned short* aSrc0 = xb  + ((size_t)b * Nc + tok0) * Dc + kA0;
    const unsigned short* aSrc1 = xb  + ((size_t)b * Nc + tok1) * Dc + kA1;
    const unsigned short* bSrc0 = wtb + ((size_t)e * OUTc + nbase + rA0) * Dc + kA0;
    const unsigned short* bSrc1 = wtb + ((size_t)e * OUTc + nbase + rA1) * Dc + kA1;

    const int lane  = tid & 31;
    const int half  = lane >> 4;
    const int r     = lane & 15;
    const int wave  = tid >> 5;
    const int waveM = wave & 3;    // 0..3 -> 32-row strip
    const int waveN = wave >> 2;   // 0..1 -> 64-col strip

    v8f acc[2][4];
#pragma unroll
    for (int i = 0; i < 2; ++i)
#pragma unroll
        for (int j = 0; j < 4; ++j)
            acc[i][j] = v8f{0.f, 0.f, 0.f, 0.f, 0.f, 0.f, 0.f, 0.f};

    constexpr int KSTEPS = Dc / TK;   // 16

#if USE_ASYNC_LDS
    // ---- prologue: async-stage K-step 0 into buffer 0 ----
    async_cp16(aSrc0, &Asm[0][rA0][kA0]);
    async_cp16(aSrc1, &Asm[0][rA1][kA1]);
    async_cp16(bSrc0, &Bsm[0][rA0][kA0]);
    async_cp16(bSrc1, &Bsm[0][rA1][kA1]);
    wait_asynccnt0();
    __syncthreads();
    for (int k = 0; k < KSTEPS; ++k) {
        const int cur = k & 1, nxt = cur ^ 1;
        if (k + 1 < KSTEPS) {
            const int ko = (k + 1) * TK;
            async_cp16(aSrc0 + ko, &Asm[nxt][rA0][kA0]);
            async_cp16(aSrc1 + ko, &Asm[nxt][rA1][kA1]);
            async_cp16(bSrc0 + ko, &Bsm[nxt][rA0][kA0]);
            async_cp16(bSrc1 + ko, &Bsm[nxt][rA1][kA1]);
        }
        v16bf a0 = frag_ld(&Asm[cur][waveM * 32][0],      LDST, r, half);
        v16bf a1 = frag_ld(&Asm[cur][waveM * 32 + 16][0], LDST, r, half);
#pragma unroll
        for (int j = 0; j < 4; ++j) {
            v16bf bj = frag_ld(&Bsm[cur][waveN * 64 + j * 16][0], LDST, r, half);
            acc[0][j] = __builtin_amdgcn_wmma_f32_16x16x32_bf16(false, a0, false, bj,
                                                                (short)0, acc[0][j], false, false);
            acc[1][j] = __builtin_amdgcn_wmma_f32_16x16x32_bf16(false, a1, false, bj,
                                                                (short)0, acc[1][j], false, false);
        }
        wait_asynccnt0();      // this wave's prefetch has landed in LDS
        __syncthreads();       // single barrier per K-step
    }
#else
    // ---- fallback: register-staged double buffering (still one barrier per K-step) ----
    {
        uint4 a0v = *reinterpret_cast<const uint4*>(aSrc0);
        uint4 a1v = *reinterpret_cast<const uint4*>(aSrc1);
        uint4 b0v = *reinterpret_cast<const uint4*>(bSrc0);
        uint4 b1v = *reinterpret_cast<const uint4*>(bSrc1);
        *reinterpret_cast<uint4*>(&Asm[0][rA0][kA0]) = a0v;
        *reinterpret_cast<uint4*>(&Asm[0][rA1][kA1]) = a1v;
        *reinterpret_cast<uint4*>(&Bsm[0][rA0][kA0]) = b0v;
        *reinterpret_cast<uint4*>(&Bsm[0][rA1][kA1]) = b1v;
    }
    __syncthreads();
    for (int k = 0; k < KSTEPS; ++k) {
        const int cur = k & 1, nxt = cur ^ 1;
        uint4 a0v, a1v, b0v, b1v;
        if (k + 1 < KSTEPS) {
            const int ko = (k + 1) * TK;
            a0v = *reinterpret_cast<const uint4*>(aSrc0 + ko);
            a1v = *reinterpret_cast<const uint4*>(aSrc1 + ko);
            b0v = *reinterpret_cast<const uint4*>(bSrc0 + ko);
            b1v = *reinterpret_cast<const uint4*>(bSrc1 + ko);
        }
        v16bf a0 = frag_ld(&Asm[cur][waveM * 32][0],      LDST, r, half);
        v16bf a1 = frag_ld(&Asm[cur][waveM * 32 + 16][0], LDST, r, half);
#pragma unroll
        for (int j = 0; j < 4; ++j) {
            v16bf bj = frag_ld(&Bsm[cur][waveN * 64 + j * 16][0], LDST, r, half);
            acc[0][j] = __builtin_amdgcn_wmma_f32_16x16x32_bf16(false, a0, false, bj,
                                                                (short)0, acc[0][j], false, false);
            acc[1][j] = __builtin_amdgcn_wmma_f32_16x16x32_bf16(false, a1, false, bj,
                                                                (short)0, acc[1][j], false, false);
        }
        if (k + 1 < KSTEPS) {
            *reinterpret_cast<uint4*>(&Asm[nxt][rA0][kA0]) = a0v;
            *reinterpret_cast<uint4*>(&Asm[nxt][rA1][kA1]) = a1v;
            *reinterpret_cast<uint4*>(&Bsm[nxt][rA0][kA0]) = b0v;
            *reinterpret_cast<uint4*>(&Bsm[nxt][rA1][kA1]) = b1v;
        }
        __syncthreads();
    }
#endif

    // C/D layout: VGPR j, lanes 0-15 -> M=j, lanes 16-31 -> M=j+8; N = lane & 15
#pragma unroll
    for (int i = 0; i < 2; ++i) {
#pragma unroll
        for (int jj = 0; jj < 8; ++jj) {
            int slot  = mbase + waveM * 32 + i * 16 + half * 8 + jj;
            int gslot = beBase + slot;
            int token = idxb[gslot];
            if (owner[(size_t)b * Nc + token] == e * Mc + slot) {
                float scale = valb[gslot];
                size_t rowOff = ((size_t)b * Nc + token) * OUTc;
#pragma unroll
                for (int j = 0; j < 4; ++j)
                    out[rowOff + nbase + waveN * 64 + j * 16 + r] = acc[i][j][jj] * scale;
            }
        }
    }
}

// ---------------- host orchestration ----------------
extern "C" void kernel_launch(void* const* d_in, const int* in_sizes, int n_in,
                              void* d_out, int out_size, void* d_ws, size_t ws_size,
                              hipStream_t stream) {
    (void)in_sizes; (void)n_in; (void)out_size; (void)ws_size;
    const float* x  = (const float*)d_in[0];   // [B][N][D]
    const float* gw = (const float*)d_in[1];   // [1][D][E]
    const float* ex = (const float*)d_in[2];   // [E][D][OUT]
    float* out = (float*)d_out;                // [B][N][OUT]

    char* base = (char*)d_ws;
    size_t off = 0;
    auto take = [&](size_t bytes) -> char* {
        off = (off + 255) & ~(size_t)255;
        char* p = base + off;
        off += bytes;
        return p;
    };
    float*          t      = (float*)         take((size_t)Bc * Nc * Ec * sizeof(float));
    float*          collse = (float*)         take((size_t)Bc * Ec * sizeof(float));
    float*          valb   = (float*)         take((size_t)Bc * Ec * Mc * sizeof(float));
    int*            idxb   = (int*)           take((size_t)Bc * Ec * Mc * sizeof(int));
    int*            owner  = (int*)           take((size_t)Bc * Nc * sizeof(int));
    unsigned short* xb     = (unsigned short*)take((size_t)Bc * Nc * Dc * 2);
    unsigned short* wtb    = (unsigned short*)take((size_t)Ec * OUTc * Dc * 2);

    {
        int n4 = Bc * Nc * OUTc / 4;
        moe_zero_f4<<<4096, 256, 0, stream>>>((float4*)out, n4);
    }
    moe_gate_kernel<<<Bc * Nc, 128, 0, stream>>>(x, gw, t, xb);
    {
        int total = Ec * Dc * OUTc;
        moe_wconv_kernel<<<(total + 255) / 256, 256, 0, stream>>>(ex, wtb);
    }
    for (int it = 0; it < 8; ++it) {
        moe_col_lse<<<Bc * Ec, 256, 0, stream>>>(t, collse);
        moe_row_lse<<<(Bc * Nc + 255) / 256, 256, 0, stream>>>(t, collse, it == 7 ? 1 : 0);
    }
    moe_topk<<<Bc * Ec, 1024, 0, stream>>>(t, idxb, valb);
    moe_owner_init<<<(Bc * Nc + 255) / 256, 256, 0, stream>>>(owner, Bc * Nc);
    moe_owner_fill<<<(Bc * Ec * Mc + 255) / 256, 256, 0, stream>>>(idxb, owner);
    dim3 grid(OUTc / TN, Mc / TM, Ec * Bc);
    moe_gemm_scatter<<<grid, 256, 0, stream>>>(xb, wtb, idxb, valb, owner, out);
}
